// MultiHeadAttention_19585050869875
// MI455X (gfx1250) — matrix operand order repaired and checked by reading, not compiled, not run on hardware
//
#include <hip/hip_runtime.h>
#include <hip/hip_bf16.h>
#include <math.h>

// Problem constants
#define B_  4
#define S_  2048
#define C_  512
#define H_  8
#define D_  64

typedef __attribute__((ext_vector_type(16))) __bf16 v16bf;
typedef __attribute__((ext_vector_type(8)))  __bf16 bf16x8;
typedef __attribute__((ext_vector_type(8)))  float  v8f;
typedef __attribute__((ext_vector_type(4)))  unsigned int u32x4;
typedef __attribute__((ext_vector_type(8)))  int i32x8;
typedef __attribute__((ext_vector_type(4)))  int i32x4;

union FragBF { v16bf v; bf16x8 h[2]; };

__device__ __forceinline__ v8f wmma_bf16(v16bf a, v16bf b, v8f c) {
  return __builtin_amdgcn_wmma_f32_16x16x32_bf16(false, a, false, b, (short)0, c,
                                                 false, false);
}

__device__ __forceinline__ bf16x8 cvt8(const float4 a, const float4 b) {
  bf16x8 r;
  r[0]=(__bf16)a.x; r[1]=(__bf16)a.y; r[2]=(__bf16)a.z; r[3]=(__bf16)a.w;
  r[4]=(__bf16)b.x; r[5]=(__bf16)b.y; r[6]=(__bf16)b.z; r[7]=(__bf16)b.w;
  return r;
}

// A-fragment (16x32 bf16): lane half h; elems 0..7 -> K=k0+8h+e ; elems 8..15 -> K=k0+16+8h+(e-8)
__device__ __forceinline__ v16bf load_a_f32(const float* rowp, int k0, int h) {
  FragBF f;
  const float4* p0 = (const float4*)(rowp + k0 + 8*h);
  f.h[0] = cvt8(p0[0], p0[1]);
  const float4* p1 = (const float4*)(rowp + k0 + 16 + 8*h);
  f.h[1] = cvt8(p1[0], p1[1]);
  return f.v;
}
__device__ __forceinline__ v16bf load_a_bf16(const __bf16* rowp, int k0, int h) {
  FragBF f;
  f.h[0] = *(const bf16x8*)(rowp + k0 + 8*h);
  f.h[1] = *(const bf16x8*)(rowp + k0 + 16 + 8*h);
  return f.v;
}
// B-fragment (32x16 bf16): lane half h; elems 0..7 -> K=k0+8h+e ; elems 8..15 -> K=k0+16+8h+(e-8)
__device__ __forceinline__ v16bf load_b_f32(const float* rowp, int k0, int h) {
  FragBF f;
  const float4* p = (const float4*)(rowp + k0 + 8*h);
  f.h[0] = cvt8(p[0], p[1]);
  const float4* p1 = (const float4*)(rowp + k0 + 16 + 8*h);
  f.h[1] = cvt8(p1[0], p1[1]);
  return f.v;
}
__device__ __forceinline__ v16bf load_b_bf16(const __bf16* rowp, int k0, int h) {
  FragBF f;
  f.h[0] = *(const bf16x8*)(rowp + k0 + 8*h);
  f.h[1] = *(const bf16x8*)(rowp + k0 + 16 + 8*h);
  return f.v;
}

// -------------------------------------------------------------------------
// Projection GEMM: out[m,n] = (sum_k X[m,k]*W[n,k] + bias[n]) * scale,
// written bf16 in [b, head, s, d] layout. Grid (M/128, C/128), block 256.
// Wave tile 64(M) x 32(N); WG tile 128x128.
// -------------------------------------------------------------------------
__global__ __launch_bounds__(256)
void gemm_qkv(const float* __restrict__ X, const float* __restrict__ W,
              const float* __restrict__ bias, __bf16* __restrict__ out,
              float scale) {
  const int tid  = threadIdx.x;
  const int wave = tid >> 5, lane = tid & 31;
  const int h = lane >> 4, r = lane & 15;
  const int wm = wave >> 2, wn = wave & 3;
  const int row0 = blockIdx.x * 128 + wm * 64;
  const int col0 = blockIdx.y * 128 + wn * 32;

  v8f acc[4][2];
#pragma unroll
  for (int i = 0; i < 4; i++)
#pragma unroll
    for (int j = 0; j < 2; j++) acc[i][j] = (v8f)0.0f;

  const float* arow[4];
#pragma unroll
  for (int i = 0; i < 4; i++) arow[i] = X + (size_t)(row0 + i*16 + r) * C_;
  const float* brow[2];
#pragma unroll
  for (int j = 0; j < 2; j++) brow[j] = W + (size_t)(col0 + j*16 + r) * C_;

  for (int k0 = 0; k0 < C_; k0 += 32) {
    v16bf bfrag[2];
#pragma unroll
    for (int j = 0; j < 2; j++) bfrag[j] = load_b_f32(brow[j], k0, h);
#pragma unroll
    for (int i = 0; i < 4; i++) {
      v16bf afrag = load_a_f32(arow[i], k0, h);
#pragma unroll
      for (int j = 0; j < 2; j++) acc[i][j] = wmma_bf16(afrag, bfrag[j], acc[i][j]);
    }
  }

#pragma unroll
  for (int i = 0; i < 4; i++) {
#pragma unroll
    for (int j = 0; j < 2; j++) {
      const int n = col0 + j*16 + r;
      const int head = n >> 6, d = n & 63;
      const float bv = bias[n];
#pragma unroll
      for (int e = 0; e < 8; e++) {
        const int m = row0 + i*16 + e + 8*h;
        const int bb = m >> 11, s = m & (S_ - 1);
        const float val = (acc[i][j][e] + bv) * scale;
        out[((size_t)(bb*H_ + head)*S_ + s)*D_ + d] = (__bf16)val;
      }
    }
  }
}

// -------------------------------------------------------------------------
// Output GEMM: out[m,n] = sum_k A[m,k]*W[n,k] + bias[n]  (A bf16, out f32)
// -------------------------------------------------------------------------
__global__ __launch_bounds__(256)
void gemm_out_k(const __bf16* __restrict__ A, const float* __restrict__ W,
                const float* __restrict__ bias, float* __restrict__ out) {
  const int tid  = threadIdx.x;
  const int wave = tid >> 5, lane = tid & 31;
  const int h = lane >> 4, r = lane & 15;
  const int wm = wave >> 2, wn = wave & 3;
  const int row0 = blockIdx.x * 128 + wm * 64;
  const int col0 = blockIdx.y * 128 + wn * 32;

  v8f acc[4][2];
#pragma unroll
  for (int i = 0; i < 4; i++)
#pragma unroll
    for (int j = 0; j < 2; j++) acc[i][j] = (v8f)0.0f;

  const __bf16* arow[4];
#pragma unroll
  for (int i = 0; i < 4; i++) arow[i] = A + (size_t)(row0 + i*16 + r) * C_;
  const float* brow[2];
#pragma unroll
  for (int j = 0; j < 2; j++) brow[j] = W + (size_t)(col0 + j*16 + r) * C_;

  for (int k0 = 0; k0 < C_; k0 += 32) {
    v16bf bfrag[2];
#pragma unroll
    for (int j = 0; j < 2; j++) bfrag[j] = load_b_f32(brow[j], k0, h);
#pragma unroll
    for (int i = 0; i < 4; i++) {
      v16bf afrag = load_a_bf16(arow[i], k0, h);
#pragma unroll
      for (int j = 0; j < 2; j++) acc[i][j] = wmma_bf16(afrag, bfrag[j], acc[i][j]);
    }
  }

#pragma unroll
  for (int i = 0; i < 4; i++) {
#pragma unroll
    for (int j = 0; j < 2; j++) {
      const int n = col0 + j*16 + r;
      const float bv = bias[n];
#pragma unroll
      for (int e = 0; e < 8; e++) {
        const int m = row0 + i*16 + e + 8*h;
        out[(size_t)m * C_ + n] = acc[i][j][e] + bv;
      }
    }
  }
}

// -------------------------------------------------------------------------
// Flash attention with bilinear-resized relative bias.
// Computes scores TRANSPOSED (keys = M, queries = N) so that the softmaxed
// probabilities in D-layout are exactly the A-fragment layout for P@V.
// K tile staged to LDS via the Tensor Data Mover (wave 0 issues, TENSORcnt).
// Grid (S/128, H, B), block 256 (8 waves x 16 queries).
// -------------------------------------------------------------------------
__global__ __launch_bounds__(256)
void attn(const __bf16* __restrict__ Q, const __bf16* __restrict__ K,
          const __bf16* __restrict__ V, const float* __restrict__ ab,
          const int* __restrict__ bidx, __bf16* __restrict__ ctx) {
  __shared__ __align__(16) float  biasT[49 * 49];
  __shared__ __align__(16) __bf16 k_lds[64 * 64];   // [key][d]
  __shared__ __align__(16) __bf16 vT_lds[64 * 64];  // [d][key]

  const int tid  = threadIdx.x;
  const int wave = tid >> 5, lane = tid & 31;
  const int h = lane >> 4, r = lane & 15;
  const int b = blockIdx.z, hd = blockIdx.y;
  const int q0 = blockIdx.x * 128 + wave * 16;

  // Gather the per-head 49x49 bias table into LDS
  for (int t = tid; t < 49 * 49; t += 256) {
    int idx = bidx[t];
    idx = idx < 0 ? 0 : (idx > 48 ? 48 : idx);
    biasT[t] = ab[hd * 49 + idx];
  }

  const size_t headbase = ((size_t)b * H_ + hd) * S_ * D_;

  // --- Tensor DMA descriptor (D#) for the 64x64 bf16 K tile -------------
  // Group 0: count=1; lds_addr[63:32]; global_addr[120:64]; type=2[127:126]
  // Group 1: data_size=1 (2B) [17:16]; tensor_dim0=64 [79:48];
  //          tensor_dim1=2048 [111:80]; tile_dim0=64 [127:112];
  //          tile_dim1=64 [143:128]; tensor_dim0_stride=64 [207:160]
  const unsigned kldsoff = (unsigned)(size_t)(void*)&k_lds[0];
  u32x4 g0;
  g0[0] = 1u;       // count = 1 valid descriptor
  g0[1] = kldsoff;  // LDS byte address
  g0[2] = 0u;       // global_addr low (per tile, set in loop)
  g0[3] = 0u;
  i32x8 g1;
  g1[0] = 0x10000;              // data_size = 1 -> 2 bytes; no mask/pad/iter
  g1[1] = (int)(64u << 16);     // tensor_dim0[15:0] = 64
  g1[2] = (int)(2048u << 16);   // tensor_dim1[15:0] = 2048
  g1[3] = (int)(64u << 16);     // tile_dim0 = 64
  g1[4] = 64;                   // tile_dim1 = 64, tile_dim2 = 0
  g1[5] = 64;                   // tensor_dim0_stride = 64
  g1[6] = 0;
  g1[7] = 0;
  i32x4 gz;
  gz[0] = 0; gz[1] = 0; gz[2] = 0; gz[3] = 0;
  i32x8 gz8;
  gz8[0] = 0; gz8[1] = 0; gz8[2] = 0; gz8[3] = 0;
  gz8[4] = 0; gz8[5] = 0; gz8[6] = 0; gz8[7] = 0;

  // Per-lane query column and its bilinear y-interpolation (fixed per lane)
  const int qglob = q0 + r;
  const float fy  = (qglob + 0.5f) * (49.0f / 2048.0f) - 0.5f;
  const float y0f = floorf(fy);
  const float wy  = fy - y0f;
  int y0 = (int)y0f, y1 = (int)y0f + 1;
  y0 = y0 < 0 ? 0 : (y0 > 48 ? 48 : y0);
  y1 = y1 < 0 ? 0 : (y1 > 48 ? 48 : y1);

  // Q fragments (B operand of scores^T WMMA), loaded once
  v16bf qfrag[2];
  {
    const __bf16* qrow = Q + headbase + (size_t)qglob * D_;
#pragma unroll
    for (int dsi = 0; dsi < 2; dsi++) qfrag[dsi] = load_b_bf16(qrow, dsi * 32, h);
  }

  v8f oacc[4];
#pragma unroll
  for (int dt = 0; dt < 4; dt++) oacc[dt] = (v8f)0.0f;
  float mrun = -3.0e38f, lrun = 0.0f;

  for (int kt0 = 0; kt0 < S_; kt0 += 64) {
    __syncthreads();  // previous tile fully consumed before overwrite

    // K tile: one TDM descriptor issued by wave 0 (EXEC ignored by TDM)
    const __bf16* kbase = K + headbase + (size_t)kt0 * D_;
    if (wave == 0) {
      const unsigned long long ga = (unsigned long long)(size_t)kbase;
      g0[2] = (unsigned)ga;
      g0[3] = ((unsigned)(ga >> 32) & 0x01FFFFFFu) | 0x80000000u;  // type=2
      __builtin_amdgcn_tensor_load_to_lds(g0, g1, gz, gz, gz8, 0);
    }

    // V tile: cooperative transposed staging (overlaps with the TDM copy)
    const __bf16* vbase = V + headbase + (size_t)kt0 * D_;
#pragma unroll
    for (int c = 0; c < 2; c++) {
      const int ch  = tid + c * 256;       // 512 chunks of 8 bf16
      const int key = ch >> 3, dc = (ch & 7) * 8;
      bf16x8 vv = *(const bf16x8*)&vbase[key * 64 + dc];
#pragma unroll
      for (int i = 0; i < 8; i++) vT_lds[(dc + i) * 64 + key] = vv[i];
    }

    if (wave == 0) __builtin_amdgcn_s_wait_tensorcnt(0);
    __syncthreads();

    // scores^T fragments: 64 keys x 16 queries
    v8f sf[4];
#pragma unroll
    for (int kt = 0; kt < 4; kt++) {
      v8f s = (v8f)0.0f;
#pragma unroll
      for (int dsi = 0; dsi < 2; dsi++) {
        v16bf kf = load_a_bf16(&k_lds[(kt * 16 + r) * 64], dsi * 32, h);
        s = wmma_bf16(kf, qfrag[dsi], s);
      }
      sf[kt] = s;
    }

    // Add bilinear bias (q is 1/sqrt(D)-prescaled), track tile max
    float tmax = -3.0e38f;
#pragma unroll
    for (int kt = 0; kt < 4; kt++) {
#pragma unroll
      for (int e = 0; e < 8; e++) {
        const int key = kt0 + kt * 16 + e + 8 * h;
        const float fx  = (key + 0.5f) * (49.0f / 2048.0f) - 0.5f;
        const float x0f = floorf(fx);
        const float wx  = fx - x0f;
        int x0 = (int)x0f, x1 = (int)x0f + 1;
        x0 = x0 < 0 ? 0 : (x0 > 48 ? 48 : x0);
        x1 = x1 < 0 ? 0 : (x1 > 48 ? 48 : x1);
        const float t00 = biasT[y0 * 49 + x0], t01 = biasT[y0 * 49 + x1];
        const float t10 = biasT[y1 * 49 + x0], t11 = biasT[y1 * 49 + x1];
        const float top = t00 + wx * (t01 - t00);
        const float bot = t10 + wx * (t11 - t10);
        const float sv = sf[kt][e] + (top + wy * (bot - top));
        sf[kt][e] = sv;
        tmax = fmaxf(tmax, sv);
      }
    }
    tmax = fmaxf(tmax, __shfl_xor(tmax, 16, 32));
    const float newm = fmaxf(mrun, tmax);
    const float corr = __expf(mrun - newm);

    float tsum = 0.0f;
#pragma unroll
    for (int kt = 0; kt < 4; kt++) {
#pragma unroll
      for (int e = 0; e < 8; e++) {
        const float p = __expf(sf[kt][e] - newm);
        sf[kt][e] = p;
        tsum += p;
      }
    }
    tsum += __shfl_xor(tsum, 16, 32);
    lrun = lrun * corr + tsum;
    mrun = newm;

    // Rescale ctx accumulators: broadcast corr from column-form to row-form
#pragma unroll
    for (int j = 0; j < 8; j++) {
      const float cj = __shfl(corr, j + 8 * h, 32);
#pragma unroll
      for (int dt = 0; dt < 4; dt++) oacc[dt][j] *= cj;
    }

    // Pack probs (D-layout of scores^T) directly into A-fragments
    v16bf pf[2];
#pragma unroll
    for (int ks = 0; ks < 2; ks++) {
      FragBF f;
#pragma unroll
      for (int e = 0; e < 8; e++) {
        f.h[0][e] = (__bf16)sf[2 * ks][e];
        f.h[1][e] = (__bf16)sf[2 * ks + 1][e];
      }
      pf[ks] = f.v;
    }

    // ctx += P @ V
#pragma unroll
    for (int dt = 0; dt < 4; dt++) {
#pragma unroll
      for (int ks = 0; ks < 2; ks++) {
        v16bf vf = load_b_bf16(&vT_lds[(dt * 16 + r) * 64], ks * 32, h);
        oacc[dt] = wmma_bf16(pf[ks], vf, oacc[dt]);
      }
    }
  }

  // Finalize: divide by row sums and store ctx in [b, s, head*D + d] bf16
  const float inv = 1.0f / lrun;
  __bf16* obase = ctx + (size_t)b * S_ * C_ + (size_t)hd * D_;
#pragma unroll
  for (int j = 0; j < 8; j++) {
    const float ij = __shfl(inv, j + 8 * h, 32);
    const int qrow = q0 + j + 8 * h;
#pragma unroll
    for (int dt = 0; dt < 4; dt++) {
      obase[(size_t)qrow * C_ + dt * 16 + r] = (__bf16)(oacc[dt][j] * ij);
    }
  }
}

// -------------------------------------------------------------------------
extern "C" void kernel_launch(void* const* d_in, const int* in_sizes, int n_in,
                              void* d_out, int out_size, void* d_ws, size_t ws_size,
                              hipStream_t stream) {
  const float* x  = (const float*)d_in[0];
  const float* Wq = (const float*)d_in[1];
  const float* bq = (const float*)d_in[2];
  const float* Wk = (const float*)d_in[3];
  const float* bk = (const float*)d_in[4];
  const float* Wv = (const float*)d_in[5];
  const float* bv = (const float*)d_in[6];
  const float* Wo = (const float*)d_in[7];
  const float* bo = (const float*)d_in[8];
  const float* ab = (const float*)d_in[9];
  const int*  bix = (const int*)d_in[10];
  float* out = (float*)d_out;

  const size_t QKV_ELEMS = (size_t)B_ * H_ * S_ * D_;  // 4,194,304
  __bf16* qb = (__bf16*)d_ws;
  __bf16* kb = qb + QKV_ELEMS;
  __bf16* vb = kb + QKV_ELEMS;
  __bf16* cb = vb + QKV_ELEMS;

  dim3 gp(B_ * S_ / 128, C_ / 128);  // (64, 4)
  gemm_qkv<<<gp, 256, 0, stream>>>(x, Wq, bq, qb, 0.125f);  // 1/sqrt(64)
  gemm_qkv<<<gp, 256, 0, stream>>>(x, Wk, bk, kb, 1.0f);
  gemm_qkv<<<gp, 256, 0, stream>>>(x, Wv, bv, vb, 1.0f);

  dim3 ga(S_ / 128, H_, B_);  // (16, 8, 4)
  attn<<<ga, 256, 0, stream>>>(qb, kb, vb, ab, bix, cb);

  gemm_out_k<<<gp, 256, 0, stream>>>(cb, Wo, bo, out);
}
